// MoE_89498528514729
// MI455X (gfx1250) — matrix operand order repaired and checked by reading, not compiled, not run on hardware
//
#include <hip/hip_runtime.h>
#include <stdint.h>

// ---------------- problem constants (match reference) ----------------
#define DIM    1024
#define INTER  512
#define NEXP   16      // routed experts
#define TOPK   4
#define NGRP   4       // expert groups
#define T_TOK  2048    // B*S
#define MTILE  32      // tokens per block (2 WMMA M-subtiles)
#define GTILE  16      // tokens per gate block

typedef __attribute__((ext_vector_type(16))) __bf16 v16bf;
typedef __attribute__((ext_vector_type(8)))  __bf16 v8bf;
typedef __attribute__((ext_vector_type(4)))  __bf16 v4bf;
typedef __attribute__((ext_vector_type(8)))  float  v8f;

static __device__ __forceinline__ v8f wmma_bf16(v16bf a, v16bf b, v8f c) {
  // D = A(16x32 bf16) * B(32x16 bf16) + C(16x16 f32)
  return __builtin_amdgcn_wmma_f32_16x16x32_bf16(false, a, false, b,
                                                 (short)0, c, false, false);
}

// ---------------- fp32 -> bf16 streaming conversion ----------------
__global__ void cvt_f32_bf16(const float* __restrict__ src,
                             __bf16* __restrict__ dst, int n4) {
  int i = blockIdx.x * blockDim.x + threadIdx.x;
  if (i >= n4) return;
  float4 v = reinterpret_cast<const float4*>(src)[i];
  v4bf o = { (__bf16)v.x, (__bf16)v.y, (__bf16)v.z, (__bf16)v.w };
  reinterpret_cast<v4bf*>(dst)[i] = o;
}

// ---------------- gate: sigmoid scores + group-limited top-k ----------------
// block = 256 threads = 16 tokens x 16 experts. Deterministic (no atomics).
__global__ __launch_bounds__(256) void gate_kernel(
    const float* __restrict__ x,        // [T, DIM]
    const float* __restrict__ gw,       // [E, DIM]
    const float* __restrict__ gb,       // [E]
    float* __restrict__ comb)           // [T, E] dense combine weights
{
  const int tl = threadIdx.x >> 4;      // token-in-block 0..15
  const int e  = threadIdx.x & 15;      // expert 0..15
  const int token = blockIdx.x * GTILE + tl;

  const float4* xr = reinterpret_cast<const float4*>(x + (size_t)token * DIM);
  const float4* wr = reinterpret_cast<const float4*>(gw + (size_t)e * DIM);
  float d = 0.f;
#pragma unroll 4
  for (int k = 0; k < DIM / 4; ++k) {
    float4 a = xr[k], b = wr[k];
    d += a.x * b.x + a.y * b.y + a.z * b.z + a.w * b.w;
  }
  const float score = 1.0f / (1.0f + __expf(-d));

  __shared__ float sc[GTILE][NEXP];
  sc[tl][e] = score;
  __syncthreads();

  if (e == 0) {
    float s[NEXP];
#pragma unroll
    for (int j = 0; j < NEXP; ++j) s[j] = sc[tl][j] + gb[j];

    // group max then top-2 groups (stable: first index wins on ties)
    float gmax[NGRP];
#pragma unroll
    for (int g = 0; g < NGRP; ++g) {
      float m = s[g * 4];
      for (int j = 1; j < 4; ++j) m = fmaxf(m, s[g * 4 + j]);
      gmax[g] = m;
    }
    int g1 = 0;
    for (int g = 1; g < NGRP; ++g) if (gmax[g] > gmax[g1]) g1 = g;
    int g2 = (g1 == 0) ? 1 : 0;
    for (int g = 0; g < NGRP; ++g)
      if (g != g1 && gmax[g] > gmax[g2]) g2 = g;

    bool keep[NEXP], used[NEXP];
#pragma unroll
    for (int j = 0; j < NEXP; ++j) {
      int g = j >> 2;
      keep[j] = (g == g1) || (g == g2);
      used[j] = false;
    }
    float row[NEXP];
#pragma unroll
    for (int j = 0; j < NEXP; ++j) row[j] = 0.f;

    for (int t = 0; t < TOPK; ++t) {     // top-4 within kept groups
      float best = -__builtin_huge_valf();
      int bi = 0;
      for (int j = 0; j < NEXP; ++j)
        if (keep[j] && !used[j] && s[j] > best) { best = s[j]; bi = j; }
      used[bi] = true;
      row[bi] = sc[tl][bi];              // ROUTE_SCALE == 1.0
    }
#pragma unroll
    for (int j = 0; j < NEXP; ++j) comb[(size_t)token * NEXP + j] = row[j];
  }
}

// ---------------- fused MoE FFN: 17 experts per 32-token tile ----------------
// block = 512 threads (16 waves), MTILE = 32 (two 16-row WMMA M-subtiles).
// Wave w: GEMM1 N-range [w*32, w*32+32), GEMM2 N-range [w*64, w*64+64).
// Each B fragment (16 contiguous bf16 per lane) feeds TWO wmmas (one per
// M-subtile) -> halves L2 B-traffic vs a 16-token tile.
__global__ __launch_bounds__(512) void moe_ffn(
    const __bf16* __restrict__ xb,    // [T, DIM]
    const __bf16* __restrict__ w1b,   // [E, INTER, DIM]
    const __bf16* __restrict__ w3b,   // [E, INTER, DIM]
    const __bf16* __restrict__ w2b,   // [E, DIM, INTER]
    const __bf16* __restrict__ ws1b,  // [INTER, DIM]
    const __bf16* __restrict__ ws3b,  // [INTER, DIM]
    const __bf16* __restrict__ ws2b,  // [DIM, INTER]
    const float*  __restrict__ comb,  // [T, E]
    float* __restrict__ out)          // [T, DIM]
{
  __shared__ __align__(16) __bf16 x_sh[MTILE * DIM];    // 64 KB
  __shared__ __align__(16) __bf16 h_sh[MTILE * INTER];  // 32 KB
  __shared__ float comb_sh[MTILE];

  const int tid  = threadIdx.x;
  const int wave = tid >> 5;                   // 0..15
  const int lane = tid & 31;
  const int base = blockIdx.x * MTILE;

  // WMMA fragment geometry (ISA 7.12.2, wave32, 16-bit)
  const int mrowA = lane & 15;                 // A-matrix row within subtile
  const int koffA = (lane < 16) ? 0 : 16;      // A K sub-offset
  const int koffB = (lane < 16) ? 0 : 16;      // B K sub-offset
  const int moff  = (lane < 16) ? 0 : 8;       // C/D row offset
  const int ncol  = lane & 15;                 // C/D column within subtile

  // stage X tile (32 rows contiguous in xb) into LDS, 128-bit chunks
  {
    const uint4* g = reinterpret_cast<const uint4*>(xb + (size_t)base * DIM);
    uint4* l = reinterpret_cast<uint4*>(x_sh);
    for (int i = tid; i < MTILE * DIM / 8; i += 512) l[i] = g[i];
  }

  v8f yacc[2][4];
#pragma unroll
  for (int mh = 0; mh < 2; ++mh)
#pragma unroll
    for (int st = 0; st < 4; ++st) yacc[mh][st] = (v8f)0.f;

  for (int e = 0; e <= NEXP; ++e) {      // 16 routed + 1 shared expert
    __syncthreads();                     // x_sh ready / previous h_sh reads done
    if (tid < MTILE)
      comb_sh[tid] = (e == NEXP) ? 1.0f : comb[(size_t)(base + tid) * NEXP + e];
    __syncthreads();

    bool active = (e == NEXP);
    if (!active)
      for (int j = 0; j < MTILE; ++j) active |= (comb_sh[j] != 0.f);
    if (!active) continue;               // block-uniform branch

    const __bf16* W1 = (e < NEXP) ? w1b + (size_t)e * INTER * DIM : ws1b;
    const __bf16* W3 = (e < NEXP) ? w3b + (size_t)e * INTER * DIM : ws3b;
    const __bf16* W2 = (e < NEXP) ? w2b + (size_t)e * DIM * INTER : ws2b;

    // ---- GEMM1: h1/h3 [32 x 32] for this wave, K = DIM ----
    v8f h1a[2][2], h3a[2][2];
#pragma unroll
    for (int mh = 0; mh < 2; ++mh)
#pragma unroll
      for (int st = 0; st < 2; ++st) { h1a[mh][st] = (v8f)0.f; h3a[mh][st] = (v8f)0.f; }

#pragma unroll 2
    for (int k0 = 0; k0 < DIM; k0 += 32) {
      v16bf a[2];
#pragma unroll
      for (int mh = 0; mh < 2; ++mh) {
        const int row = (mh * 16 + mrowA) * DIM + k0 + koffA;
        v8bf alo = *reinterpret_cast<const v8bf*>(&x_sh[row]);
        v8bf ahi = *reinterpret_cast<const v8bf*>(&x_sh[row + 8]);
        a[mh] = __builtin_shufflevector(alo, ahi, 0,1,2,3,4,5,6,7,
                                                  8,9,10,11,12,13,14,15);
      }
#pragma unroll
      for (int st = 0; st < 2; ++st) {
        const int n = wave * 32 + st * 16 + ncol;       // inter index 0..511
        const size_t boff = (size_t)n * DIM + k0 + koffB;
        v16bf b1 = *reinterpret_cast<const v16bf*>(W1 + boff);
        h1a[0][st] = wmma_bf16(a[0], b1, h1a[0][st]);
        h1a[1][st] = wmma_bf16(a[1], b1, h1a[1][st]);
        v16bf b3 = *reinterpret_cast<const v16bf*>(W3 + boff);
        h3a[0][st] = wmma_bf16(a[0], b3, h3a[0][st]);
        h3a[1][st] = wmma_bf16(a[1], b3, h3a[1][st]);
      }
    }

    // ---- epilogue: h = silu(h1) * h3 * gate_weight -> LDS bf16 ----
#pragma unroll
    for (int mh = 0; mh < 2; ++mh) {
#pragma unroll
      for (int r = 0; r < 8; ++r) {
        const int m = mh * 16 + r + moff;
        const float cw = comb_sh[m];
#pragma unroll
        for (int st = 0; st < 2; ++st) {
          const float v1 = h1a[mh][st][r];
          const float hv = (v1 / (1.0f + __expf(-v1))) * h3a[mh][st][r] * cw;
          h_sh[m * INTER + wave * 32 + st * 16 + ncol] = (__bf16)hv;
        }
      }
    }
    __syncthreads();                     // h tile complete

    // ---- GEMM2: y[32 x 64] += h[32 x 512] * W2[n][k], K = INTER ----
#pragma unroll 2
    for (int k0 = 0; k0 < INTER; k0 += 32) {
      v16bf a[2];
#pragma unroll
      for (int mh = 0; mh < 2; ++mh) {
        const int row = (mh * 16 + mrowA) * INTER + k0 + koffA;
        v8bf alo = *reinterpret_cast<const v8bf*>(&h_sh[row]);
        v8bf ahi = *reinterpret_cast<const v8bf*>(&h_sh[row + 8]);
        a[mh] = __builtin_shufflevector(alo, ahi, 0,1,2,3,4,5,6,7,
                                                  8,9,10,11,12,13,14,15);
      }
#pragma unroll
      for (int st = 0; st < 4; ++st) {
        const int n = wave * 64 + st * 16 + ncol;       // out dim 0..1023
        v16bf b = *reinterpret_cast<const v16bf*>(W2 + (size_t)n * INTER + k0 + koffB);
        yacc[0][st] = wmma_bf16(a[0], b, yacc[0][st]);
        yacc[1][st] = wmma_bf16(a[1], b, yacc[1][st]);
      }
    }
  }

  // ---- store y (routed + shared already summed in accumulators) ----
#pragma unroll
  for (int mh = 0; mh < 2; ++mh) {
#pragma unroll
    for (int st = 0; st < 4; ++st) {
#pragma unroll
      for (int r = 0; r < 8; ++r) {
        const int m = mh * 16 + r + moff;
        const int col = wave * 64 + st * 16 + ncol;
        out[(size_t)(base + m) * DIM + col] = yacc[mh][st][r];
      }
    }
  }
}

// ---------------- host-side launch ----------------
extern "C" void kernel_launch(void* const* d_in, const int* in_sizes, int n_in,
                              void* d_out, int out_size, void* d_ws, size_t ws_size,
                              hipStream_t stream) {
  (void)in_sizes; (void)n_in; (void)out_size; (void)ws_size;

  const float* x    = (const float*)d_in[0];  // [2,1024,1024]
  const float* gw   = (const float*)d_in[1];  // [16,1024]
  const float* gb   = (const float*)d_in[2];  // [16]
  const float* w1   = (const float*)d_in[3];  // [16,512,1024]
  const float* w2   = (const float*)d_in[4];  // [16,1024,512]
  const float* w3   = (const float*)d_in[5];  // [16,512,1024]
  const float* ws1  = (const float*)d_in[6];  // [512,1024]
  const float* ws2  = (const float*)d_in[7];  // [1024,512]
  const float* ws3  = (const float*)d_in[8];  // [512,1024]
  float* out = (float*)d_out;

  // workspace layout (bf16 weights are L2-resident: ~55 MB total)
  char* ws = (char*)d_ws;
  const size_t SZ_XB  = (size_t)T_TOK * DIM * 2;
  const size_t SZ_W   = (size_t)NEXP * INTER * DIM * 2;   // w1/w3/w2 each
  const size_t SZ_WS  = (size_t)INTER * DIM * 2;
  __bf16* xb   = (__bf16*)(ws);
  __bf16* w1b  = (__bf16*)(ws + SZ_XB);
  __bf16* w3b  = (__bf16*)(ws + SZ_XB + SZ_W);
  __bf16* w2b  = (__bf16*)(ws + SZ_XB + 2 * SZ_W);
  __bf16* ws1b = (__bf16*)(ws + SZ_XB + 3 * SZ_W);
  __bf16* ws3b = (__bf16*)(ws + SZ_XB + 3 * SZ_W + SZ_WS);
  __bf16* ws2b = (__bf16*)(ws + SZ_XB + 3 * SZ_W + 2 * SZ_WS);
  float*  comb = (float*) (ws + SZ_XB + 3 * SZ_W + 3 * SZ_WS);

  auto cvt = [&](const float* s, __bf16* d, size_t n) {
    int n4 = (int)(n / 4);
    cvt_f32_bf16<<<(n4 + 255) / 256, 256, 0, stream>>>(s, d, n4);
  };
  cvt(x,   xb,   (size_t)T_TOK * DIM);
  cvt(w1,  w1b,  (size_t)NEXP * INTER * DIM);
  cvt(w3,  w3b,  (size_t)NEXP * INTER * DIM);
  cvt(w2,  w2b,  (size_t)NEXP * DIM * INTER);
  cvt(ws1, ws1b, (size_t)INTER * DIM);
  cvt(ws3, ws3b, (size_t)INTER * DIM);
  cvt(ws2, ws2b, (size_t)DIM * INTER);

  gate_kernel<<<T_TOK / GTILE, 256, 0, stream>>>(x, gw, gb, comb);

  moe_ffn<<<T_TOK / MTILE, 512, 0, stream>>>(xb, w1b, w3b, w2b,
                                             ws1b, ws3b, ws2b, comb, out);
}
